// MultiHeadAttention_38835094290990
// MI455X (gfx1250) — matrix operand order repaired and checked by reading, not compiled, not run on hardware
//
#include <hip/hip_runtime.h>
#include <hip/hip_bf16.h>
#include <stdint.h>

// ---------------- problem constants ----------------
constexpr int N_ = 4;
constexpr int T_ = 2048;
constexpr int D_ = 1024;
constexpr int H_ = 16;
constexpr int DH_ = 64;
constexpr int HD_ = H_ * DH_;      // 1024
constexpr int S_ = N_ * T_;        // 8192 rows
constexpr float CAP_ = 30.0f;
constexpr float EPS_ = 1e-5f;

typedef __bf16 bf16_t;
typedef __attribute__((ext_vector_type(16))) __bf16 v16bf;
typedef __attribute__((ext_vector_type(8)))  __bf16 v8bf;
typedef __attribute__((ext_vector_type(8)))  float  v8f;

// ---------------- WMMA / fragment helpers ----------------
__device__ __forceinline__ v8f wmma_bf16(v16bf a, v16bf b, v8f c) {
  return __builtin_amdgcn_wmma_f32_16x16x32_bf16(
      false, a, false, b, (short)0, c, false, false);
}

// A-fragment (16x32, M x K): lane(0-15)=row; half=0 -> K{0..7,16..23},
// half=1 -> K{8..15,24..31}.
__device__ __forceinline__ v16bf ld_a(const bf16_t* base, int stride) {
  const int lane = threadIdx.x & 31;
  const bf16_t* p = base + (size_t)(lane & 15) * stride + ((lane >> 4) << 3);
  v8bf lo = *(const v8bf*)p;
  v8bf hi = *(const v8bf*)(p + 16);
  v16bf r;
#pragma unroll
  for (int i = 0; i < 8; ++i) { r[i] = lo[i]; r[i + 8] = hi[i]; }
  return r;
}

// B-fragment (32x16, K x N): lane(0-15)=col; half=0 -> K{0..7,16..23}, half=1 -> K{8..15,24..31}.
// base at (K0, col0), column-major (stride between columns).
__device__ __forceinline__ v16bf ld_b(const bf16_t* base, int stride) {
  const int lane = threadIdx.x & 31;
  const bf16_t* p = base + (size_t)(lane & 15) * stride + ((lane >> 4) << 3);
  v8bf lo = *(const v8bf*)p;
  v8bf hi = *(const v8bf*)(p + 16);
  v16bf r;
#pragma unroll
  for (int i = 0; i < 8; ++i) { r[i] = lo[i]; r[i + 8] = hi[i]; }
  return r;
}

// ---------------- CDNA5 async global -> LDS DMA ----------------
__device__ __forceinline__ void async_to_lds_b128(const void* lds_p, const void* gp) {
  const uint32_t lds32 = (uint32_t)(uintptr_t)lds_p;   // LDS address is the pointer's low 32 bits
  const uint64_t ga = (uint64_t)(uintptr_t)gp;
  asm volatile("global_load_async_to_lds_b128 %0, %1, off"
               :: "v"(lds32), "v"(ga) : "memory");
}
__device__ __forceinline__ void wait_async0() {
  asm volatile("s_wait_asynccnt 0x0" ::: "memory");
}

// ---------------- shared GEMM building blocks (tile M=128, N=64, K-step 64) ----
// A tile: 128 x 64 bf16 (padded rows of 72), filled by async DMA, 4 chunks/thread
__device__ __forceinline__ void issue_a_tile(const bf16_t* __restrict__ gsrc,
                                             int row_stride, bf16_t (*As)[72]) {
  const int tid = threadIdx.x;
#pragma unroll
  for (int j = 0; j < 4; ++j) {
    const int c = tid + j * 256;
    const int row = c >> 3, sub = c & 7;
    async_to_lds_b128(&As[row][sub * 8],
                      gsrc + (size_t)row * row_stride + sub * 8);
  }
}
// W tile: 64(K) x 64(N) fp32, staged through registers (conversion + transpose)
__device__ __forceinline__ void load_w_regs(const float* __restrict__ W, int ldn,
                                            int k0, int n0, float4* wr) {
  const int tid = threadIdx.x;
#pragma unroll
  for (int j = 0; j < 4; ++j) {
    const int c = tid + j * 256;
    const int kk = c >> 4, ns = c & 15;
    wr[j] = *(const float4*)&W[(size_t)(k0 + kk) * ldn + n0 + ns * 4];
  }
}
__device__ __forceinline__ void store_w_regs(const float4* wr, bf16_t (*Bs)[72]) {
  const int tid = threadIdx.x;
#pragma unroll
  for (int j = 0; j < 4; ++j) {
    const int c = tid + j * 256;
    const int kk = c >> 4, ns = c & 15;
    Bs[ns * 4 + 0][kk] = (bf16_t)wr[j].x;
    Bs[ns * 4 + 1][kk] = (bf16_t)wr[j].y;
    Bs[ns * 4 + 2][kk] = (bf16_t)wr[j].z;
    Bs[ns * 4 + 3][kk] = (bf16_t)wr[j].w;
  }
}
// 8 WMMAs per wave per K-step (2 K-chunks x 4 n-tiles)
__device__ __forceinline__ void gemm_step(const bf16_t (*As)[72],
                                          const bf16_t (*Bs)[72],
                                          int wid, v8f* acc) {
#pragma unroll
  for (int c = 0; c < 2; ++c) {
    const v16bf af = ld_a(&As[wid * 16][c * 32], 72);
#pragma unroll
    for (int nt = 0; nt < 4; ++nt) {
      const v16bf bf = ld_b(&Bs[nt * 16][c * 32], 72);
      acc[nt] = wmma_bf16(af, bf, acc[nt]);
    }
  }
}

// ---------------- 1) LayerNorm -> bf16 ----------------
__global__ __launch_bounds__(256) void mha_ln_kernel(
    const float* __restrict__ x, const float* __restrict__ g,
    const float* __restrict__ b, bf16_t* __restrict__ hb) {
  __shared__ float red[256];
  const int tid = threadIdx.x;
  const int row = blockIdx.x;
  const float4 xv = ((const float4*)(x + (size_t)row * D_))[tid];

  red[tid] = xv.x + xv.y + xv.z + xv.w;
  __syncthreads();
  for (int off = 128; off > 0; off >>= 1) {
    if (tid < off) red[tid] += red[tid + off];
    __syncthreads();
  }
  const float mean = red[0] * (1.0f / D_);
  __syncthreads();

  const float dx = xv.x - mean, dy = xv.y - mean, dz = xv.z - mean, dw = xv.w - mean;
  red[tid] = dx * dx + dy * dy + dz * dz + dw * dw;
  __syncthreads();
  for (int off = 128; off > 0; off >>= 1) {
    if (tid < off) red[tid] += red[tid + off];
    __syncthreads();
  }
  const float inv = rsqrtf(red[0] * (1.0f / D_) + EPS_);

  const float4 gv = ((const float4*)g)[tid];
  const float4 bv = ((const float4*)b)[tid];
  bf16_t* hr = hb + (size_t)row * D_ + tid * 4;
  hr[0] = (bf16_t)(dx * inv * gv.x + bv.x);
  hr[1] = (bf16_t)(dy * inv * gv.y + bv.y);
  hr[2] = (bf16_t)(dz * inv * gv.z + bv.z);
  hr[3] = (bf16_t)(dw * inv * gv.w + bv.w);
}

// ---------------- 2) QKV GEMM + bias + RoPE + head scatter ----------------
// grid: (16 n-tiles == head, 64 m-tiles, z = 0/1/2 for q/k/v)
__global__ __launch_bounds__(256) void mha_qkv_kernel(
    const bf16_t* __restrict__ hb,
    const float* __restrict__ Wq, const float* __restrict__ bq,
    const float* __restrict__ Wk, const float* __restrict__ bk,
    const float* __restrict__ Wv, const float* __restrict__ bv,
    bf16_t* __restrict__ qr, bf16_t* __restrict__ kr, bf16_t* __restrict__ vt) {
  __shared__ bf16_t As[2][128][72];   // double-buffered A (async DMA)
  __shared__ bf16_t Bs[2][64][72];    // double-buffered W^T bf16
  __shared__ float  st[8][16][64];    // per-wave 16x64 f32 stage

  const int tid = threadIdx.x;
  const int wid = tid >> 5, lane = tid & 31;
  const int half = lane >> 4, l15 = lane & 15;
  const int z = blockIdx.z;
  const int head = blockIdx.x;
  const int n0 = head * 64;
  const int m0 = blockIdx.y * 128;

  const float* W    = (z == 0) ? Wq : (z == 1) ? Wk : Wv;
  const float* bias = (z == 0) ? bq : (z == 1) ? bk : bv;
  const bf16_t* a0 = hb + (size_t)m0 * D_;

  v8f acc[4];
#pragma unroll
  for (int nt = 0; nt < 4; ++nt)
#pragma unroll
    for (int r = 0; r < 8; ++r) acc[nt][r] = 0.0f;

  // prologue: tile 0
  issue_a_tile(a0, D_, As[0]);
  float4 wr[4];
  load_w_regs(W, HD_, 0, n0, wr);
  store_w_regs(wr, Bs[0]);
  wait_async0();
  __syncthreads();

  constexpr int NSTEP = D_ / 64;      // 16
  for (int it = 0; it < NSTEP; ++it) {
    const int cur = it & 1, nxt = cur ^ 1;
    const bool more = (it + 1) < NSTEP;
    if (more) {
      issue_a_tile(a0 + (it + 1) * 64, D_, As[nxt]);        // DMA next A
      load_w_regs(W, HD_, (it + 1) * 64, n0, wr);           // next W to regs
      if (it + 2 < NSTEP)
        __builtin_prefetch(&W[(size_t)(it + 2) * 64 * HD_ + n0 + lane], 0, 0);
    }
    gemm_step(As[cur], Bs[cur], wid, acc);                  // 8 WMMAs
    if (more) {
      store_w_regs(wr, Bs[nxt]);
      wait_async0();
      __syncthreads();
    }
  }

  // D-fragment -> LDS stage with bias (lane=col, vgpr r=row, +8 upper half)
#pragma unroll
  for (int nt = 0; nt < 4; ++nt)
#pragma unroll
    for (int r = 0; r < 8; ++r)
      st[wid][r + half * 8][nt * 16 + l15] = acc[nt][r] + bias[n0 + nt * 16 + l15];

  if (z < 2) {  // RoPE for q / k -> [b,h,t,dh] bf16
    bf16_t* dst = (z == 0) ? qr : kr;
    for (int j = lane; j < 16 * 32; j += 32) {
      const int row = j >> 5, pair = j & 31;
      const int gm = m0 + wid * 16 + row;
      const int bb = gm / T_, t = gm - bb * T_;
      const float invf = __powf(10000.0f, -(float)pair * (1.0f / 32.0f));
      float sn, cs;
      __sincosf((float)t * invf, &sn, &cs);
      const float x1 = st[wid][row][2 * pair];
      const float x2 = st[wid][row][2 * pair + 1];
      const size_t o = ((size_t)(bb * H_ + head) * T_ + t) * DH_ + 2 * pair;
      dst[o]     = (bf16_t)(x1 * cs - x2 * sn);
      dst[o + 1] = (bf16_t)(x1 * sn + x2 * cs);
    }
  } else {      // V -> [b,h,dh,t] bf16 (pre-transposed for P*V B-fragments)
    for (int e = lane; e < 16 * 64; e += 32) {
      const int row = e >> 6, dh = e & 63;
      const int gm = m0 + wid * 16 + row;
      const int bb = gm / T_, t = gm - bb * T_;
      vt[((size_t)(bb * H_ + head) * DH_ + dh) * T_ + t] = (bf16_t)st[wid][row][dh];
    }
  }
}

// ---------------- 3) flash attention ----------------
// grid: (T/64, H, N); block = 128 (4 waves x 16 query rows)
__global__ __launch_bounds__(128) void mha_attn_kernel(
    const bf16_t* __restrict__ qr, const bf16_t* __restrict__ kr,
    const bf16_t* __restrict__ vt, const int* __restrict__ lengths,
    bf16_t* __restrict__ ao) {
  __shared__ bf16_t pst[4][16][72];   // per-wave P stage (D-layout -> A-layout)

  const int wid = threadIdx.x >> 5, lane = threadIdx.x & 31;
  const int half = lane >> 4, l15 = lane & 15;
  const int bb = blockIdx.z, h = blockIdx.y;
  const int t0 = blockIdx.x * 64 + wid * 16;
  const size_t bh = (size_t)(bb * H_ + h);
  const bf16_t* qp = qr + (bh * T_ + t0) * DH_;
  const bf16_t* kp = kr + bh * T_ * DH_;
  const bf16_t* vp = vt + bh * DH_ * T_;
  const int len = lengths[bb];

  const v16bf qa0 = ld_a(qp, DH_);
  const v16bf qa1 = ld_a(qp + 32, DH_);

  float m_i[8], l_i[8];
  v8f o[4];
#pragma unroll
  for (int r = 0; r < 8; ++r) { m_i[r] = -3.0e38f; l_i[r] = 0.0f; }
#pragma unroll
  for (int nt = 0; nt < 4; ++nt)
#pragma unroll
    for (int r = 0; r < 8; ++r) o[nt][r] = 0.0f;

  int kend = t0 + 16;
  if (len < kend) kend = len;

  for (int kb = 0; kb < kend; kb += 64) {
    if (kb + 64 < kend) {             // prefetch next K/V blocks
      __builtin_prefetch(kp + (size_t)(kb + 64 + lane * 2) * DH_, 0, 0);
      __builtin_prefetch(vp + (size_t)(lane * 2) * T_ + kb + 64, 0, 0);
    }
    // --- S = Q K^T over a 64-key block ---
    v8f sf[4];
#pragma unroll
    for (int kt = 0; kt < 4; ++kt) {
      const bf16_t* kbase = kp + (size_t)(kb + kt * 16) * DH_;
      v8f zacc;
#pragma unroll
      for (int r = 0; r < 8; ++r) zacc[r] = 0.0f;
      zacc = wmma_bf16(qa0, ld_b(kbase, DH_), zacc);
      zacc = wmma_bf16(qa1, ld_b(kbase + 32, DH_), zacc);
      sf[kt] = zacc;
    }

    // --- scale, tanh-cap, mask, online softmax (stats within 16-lane half) ---
#pragma unroll
    for (int r = 0; r < 8; ++r) {
      const int t = t0 + r + half * 8;
      float sv[4];
      float rowmax = -3.0e38f;
#pragma unroll
      for (int kt = 0; kt < 4; ++kt) {
        const int key = kb + kt * 16 + l15;
        float s = sf[kt][r] * 0.125f;
        s = CAP_ * tanhf(s * (1.0f / CAP_));
        if (key > t || key >= len) s = -3.0e38f;
        sv[kt] = s;
        rowmax = fmaxf(rowmax, s);
      }
      rowmax = fmaxf(rowmax, __shfl_xor(rowmax, 1));
      rowmax = fmaxf(rowmax, __shfl_xor(rowmax, 2));
      rowmax = fmaxf(rowmax, __shfl_xor(rowmax, 4));
      rowmax = fmaxf(rowmax, __shfl_xor(rowmax, 8));
      const float mnew = fmaxf(m_i[r], rowmax);
      const float alpha = __expf(m_i[r] - mnew);
      float rs = 0.0f;
#pragma unroll
      for (int kt = 0; kt < 4; ++kt) {
        const float p = __expf(sv[kt] - mnew);
        rs += p;
        pst[wid][r + half * 8][kt * 16 + l15] = (bf16_t)p;
      }
      rs += __shfl_xor(rs, 1);
      rs += __shfl_xor(rs, 2);
      rs += __shfl_xor(rs, 4);
      rs += __shfl_xor(rs, 8);
      l_i[r] = l_i[r] * alpha + rs;
      m_i[r] = mnew;
#pragma unroll
      for (int nt = 0; nt < 4; ++nt) o[nt][r] *= alpha;
    }

    // --- O += P V (per-wave LDS slice; same-wave DS ops in-order) ---
    const v16bf pa0 = ld_a(&pst[wid][0][0], 72);
    const v16bf pa1 = ld_a(&pst[wid][0][32], 72);
#pragma unroll
    for (int nt = 0; nt < 4; ++nt) {
      const bf16_t* vbase = vp + (size_t)(nt * 16) * T_ + kb;
      v8f t_acc = o[nt];
      t_acc = wmma_bf16(pa0, ld_b(vbase, T_), t_acc);
      t_acc = wmma_bf16(pa1, ld_b(vbase + 32, T_), t_acc);
      o[nt] = t_acc;
    }
  }

#pragma unroll
  for (int nt = 0; nt < 4; ++nt)
#pragma unroll
    for (int r = 0; r < 8; ++r) {
      const int t = t0 + r + half * 8;
      const float val = o[nt][r] / l_i[r];
      ao[((size_t)bb * T_ + t) * HD_ + h * DH_ + nt * 16 + l15] = (bf16_t)val;
    }
}

// ---------------- 4) output projection GEMM (fp32 out + bias) ----------------
__global__ __launch_bounds__(256) void mha_out_kernel(
    const bf16_t* __restrict__ ao, const float* __restrict__ Wo,
    const float* __restrict__ bo, float* __restrict__ out) {
  __shared__ bf16_t As[2][128][72];
  __shared__ bf16_t Bs[2][64][72];

  const int tid = threadIdx.x;
  const int wid = tid >> 5, lane = tid & 31;
  const int half = lane >> 4, l15 = lane & 15;
  const int n0 = blockIdx.x * 64;
  const int m0 = blockIdx.y * 128;
  const bf16_t* a0 = ao + (size_t)m0 * HD_;

  v8f acc[4];
#pragma unroll
  for (int nt = 0; nt < 4; ++nt)
#pragma unroll
    for (int r = 0; r < 8; ++r) acc[nt][r] = 0.0f;

  issue_a_tile(a0, HD_, As[0]);
  float4 wr[4];
  load_w_regs(Wo, D_, 0, n0, wr);
  store_w_regs(wr, Bs[0]);
  wait_async0();
  __syncthreads();

  constexpr int NSTEP = HD_ / 64;     // 16
  for (int it = 0; it < NSTEP; ++it) {
    const int cur = it & 1, nxt = cur ^ 1;
    const bool more = (it + 1) < NSTEP;
    if (more) {
      issue_a_tile(a0 + (it + 1) * 64, HD_, As[nxt]);
      load_w_regs(Wo, D_, (it + 1) * 64, n0, wr);
      if (it + 2 < NSTEP)
        __builtin_prefetch(&Wo[(size_t)(it + 2) * 64 * D_ + n0 + lane], 0, 0);
    }
    gemm_step(As[cur], Bs[cur], wid, acc);
    if (more) {
      store_w_regs(wr, Bs[nxt]);
      wait_async0();
      __syncthreads();
    }
  }

#pragma unroll
  for (int nt = 0; nt < 4; ++nt)
#pragma unroll
    for (int r = 0; r < 8; ++r) {
      const int gm = m0 + wid * 16 + r + half * 8;
      const int col = n0 + nt * 16 + l15;
      out[(size_t)gm * D_ + col] = acc[nt][r] + bo[col];
    }
}

// ---------------- host launch ----------------
extern "C" void kernel_launch(void* const* d_in, const int* in_sizes, int n_in,
                              void* d_out, int out_size, void* d_ws, size_t ws_size,
                              hipStream_t stream) {
  (void)in_sizes; (void)n_in; (void)out_size; (void)ws_size;

  const float* x    = (const float*)d_in[0];
  const float* ln_g = (const float*)d_in[1];
  const float* ln_b = (const float*)d_in[2];
  const float* Wq   = (const float*)d_in[3];
  const float* bq   = (const float*)d_in[4];
  const float* Wk   = (const float*)d_in[5];
  const float* bk   = (const float*)d_in[6];
  const float* Wv   = (const float*)d_in[7];
  const float* bv   = (const float*)d_in[8];
  const float* Wo   = (const float*)d_in[9];
  const float* bo   = (const float*)d_in[10];
  const int* lengths = (const int*)d_in[13];   // masks rebuilt from lengths
  float* out = (float*)d_out;

  const size_t SD = (size_t)S_ * 1024;
  bf16_t* hb = (bf16_t*)d_ws;        // LN output,      [S, D]
  bf16_t* qr = hb + SD;              // Q after RoPE,   [b,h,t,dh]
  bf16_t* kr = qr + SD;              // K after RoPE,   [b,h,t,dh]
  bf16_t* vt = kr + SD;              // V transposed,   [b,h,dh,t]
  bf16_t* ao = vt + SD;              // attention out,  [S, H*DH]

  mha_ln_kernel<<<S_, 256, 0, stream>>>(x, ln_g, ln_b, hb);
  mha_qkv_kernel<<<dim3(HD_ / 64, S_ / 128, 3), 256, 0, stream>>>(
      hb, Wq, bq, Wk, bk, Wv, bv, qr, kr, vt);
  mha_attn_kernel<<<dim3(T_ / 64, H_, N_), 128, 0, stream>>>(
      qr, kr, vt, lengths, ao);
  mha_out_kernel<<<dim3(D_ / 64, S_ / 128), 256, 0, stream>>>(ao, Wo, bo, out);
}
